// CustomBERTModel_64287070486553
// MI455X (gfx1250) — compile-verified, matrix operand order checked by reading, not run on hardware
//
#include <hip/hip_runtime.h>

typedef __attribute__((ext_vector_type(16))) _Float16 v16h;
typedef __attribute__((ext_vector_type(8)))  float    v8f;
typedef __attribute__((ext_vector_type(4)))  float    vf4;   // native vec for NT b128 ops

#define KTOP 16

// ---------------- small branchless building blocks (all register-resident) ----

__device__ __forceinline__ void ce_desc(float& va, int& ia, float& vb, int& ib) {
  // compare-exchange: keep larger in (va,ia)
  const bool sw = vb > va;
  const float lo = sw ? va : vb;
  const float hi = sw ? vb : va;
  const int  ilo = sw ? ia : ib;
  const int  ihi = sw ? ib : ia;
  va = hi; ia = ihi; vb = lo; ib = ilo;
}

// insert x into sorted-descending 16-list; caller guarantees x > v[15]
__device__ __forceinline__ void insert_desc(float (&v)[KTOP], int (&id)[KTOP],
                                            float x, int xi) {
#pragma unroll
  for (int j = KTOP - 1; j >= 1; --j) {
    const bool gp = x > v[j - 1];
    const bool gc = x > v[j];
    const float nv = gp ? v[j - 1] : (gc ? x : v[j]);
    const int   ni = gp ? id[j - 1] : (gc ? xi : id[j]);
    v[j] = nv; id[j] = ni;
  }
  const bool g0 = x > v[0];
  id[0] = g0 ? xi : id[0];
  v[0]  = g0 ? x  : v[0];
}

// merge two sorted-descending 16-lists, keep top-16 sorted descending in (v,id)
__device__ __forceinline__ void merge_desc(float (&v)[KTOP], int (&id)[KTOP],
                                           const float (&w)[KTOP], const int (&wi)[KTOP]) {
  float m[KTOP]; int mi[KTOP];
#pragma unroll
  for (int j = 0; j < KTOP; ++j) {          // bitonic "half-cleaner": max(a[j], b[15-j])
    const bool t = w[KTOP - 1 - j] > v[j];
    m[j]  = t ? w[KTOP - 1 - j]  : v[j];
    mi[j] = t ? wi[KTOP - 1 - j] : id[j];
  }
#pragma unroll
  for (int d = 8; d >= 1; d >>= 1) {        // bitonic merge of 16 -> sorted desc
#pragma unroll
    for (int j = 0; j < KTOP; ++j) {
      if ((j & d) == 0) ce_desc(m[j], mi[j], m[j + d], mi[j + d]);
    }
  }
#pragma unroll
  for (int j = 0; j < KTOP; ++j) { v[j] = m[j]; id[j] = mi[j]; }
}

// butterfly merge across all 32 lanes of a wave (ds_permute traffic only)
__device__ __forceinline__ void wave_merge(float (&v)[KTOP], int (&id)[KTOP]) {
#pragma unroll
  for (int off = 1; off < 32; off <<= 1) {
    float w[KTOP]; int wi[KTOP];
#pragma unroll
    for (int j = 0; j < KTOP; ++j) {
      w[j]  = __shfl_xor(v[j],  off, 32);
      wi[j] = __shfl_xor(id[j], off, 32);
    }
    merge_desc(v, id, w, wi);
  }
}

// ------- kernel 1: zero the dense output (NT b128 store stream, no L2 churn) --

__global__ void zero_out_kernel(float* __restrict__ out, size_t n) {
  const size_t n4 = n >> 2;
  size_t i = (size_t)blockIdx.x * blockDim.x + threadIdx.x;
  const size_t stride = (size_t)gridDim.x * blockDim.x;
  const vf4 z = {0.f, 0.f, 0.f, 0.f};
  vf4* o4 = (vf4*)out;
  for (; i < n4; i += stride) __builtin_nontemporal_store(z, &o4[i]);
  if (blockIdx.x == 0 && threadIdx.x < (n & 3)) out[(n4 << 2) + threadIdx.x] = 0.f;
}

// ------- kernel 2: per-row top-16 (one 256-thread WG per row, NT b128 loads) --

__global__ void __launch_bounds__(256)
topk_rows_kernel(const float* __restrict__ logits,
                 float* __restrict__ out_vals, int* __restrict__ out_ids, int V) {
  const int row  = blockIdx.x;
  const float* rowp = logits + (size_t)row * (size_t)V;
  const int t    = threadIdx.x;
  const int lane = t & 31;
  const int wave = t >> 5;

  float v[KTOP]; int id[KTOP];
#pragma unroll
  for (int j = 0; j < KTOP; ++j) { v[j] = -__builtin_inff(); id[j] = 0; }

  const int V4 = V >> 2;
  const vf4* row4 = (const vf4*)rowp;
  for (int i = t; i < V4; i += 256) {
    const vf4 d = __builtin_nontemporal_load(&row4[i]);   // single-use stream: TH=NT
    const int base = i << 2;
    if (d.x > v[KTOP - 1]) insert_desc(v, id, d.x, base + 0);
    if (d.y > v[KTOP - 1]) insert_desc(v, id, d.y, base + 1);
    if (d.z > v[KTOP - 1]) insert_desc(v, id, d.z, base + 2);
    if (d.w > v[KTOP - 1]) insert_desc(v, id, d.w, base + 3);
  }
  for (int i = (V4 << 2) + t; i < V; i += 256) {   // tail (empty for V=32000)
    const float x = rowp[i];
    if (x > v[KTOP - 1]) insert_desc(v, id, x, i);
  }

  wave_merge(v, id);                               // lane0 of each wave: wave top-16

  __shared__ float sv[8 * KTOP];
  __shared__ int   si[8 * KTOP];
  if (lane == 0) {
#pragma unroll
    for (int j = 0; j < KTOP; ++j) { sv[wave * KTOP + j] = v[j]; si[wave * KTOP + j] = id[j]; }
  }
  __syncthreads();

  if (wave == 0) {                                 // reduce 8*16 = 128 candidates
    float c[4]; int ci[4];
#pragma unroll
    for (int j = 0; j < 4; ++j) { c[j] = sv[lane * 4 + j]; ci[j] = si[lane * 4 + j]; }
    ce_desc(c[0], ci[0], c[2], ci[2]);             // sort-4 descending
    ce_desc(c[1], ci[1], c[3], ci[3]);
    ce_desc(c[0], ci[0], c[1], ci[1]);
    ce_desc(c[2], ci[2], c[3], ci[3]);
    ce_desc(c[1], ci[1], c[2], ci[2]);
    float f[KTOP]; int fi[KTOP];
    f[0] = c[0]; fi[0] = ci[0];
    f[1] = c[1]; fi[1] = ci[1];
    f[2] = c[2]; fi[2] = ci[2];
    f[3] = c[3]; fi[3] = ci[3];
#pragma unroll
    for (int j = 4; j < KTOP; ++j) { f[j] = -__builtin_inff(); fi[j] = 0; }
    wave_merge(f, fi);
    if (lane == 0) {
#pragma unroll
      for (int j = 0; j < KTOP; ++j) {
        out_vals[row * KTOP + j] = f[j];
        out_ids [row * KTOP + j] = fi[j];
      }
    }
  }
}

// ------- kernel 3: WMMA 16x16 linear + softmax + masked scatter (1 wave/16 rows)

__global__ void __launch_bounds__(32)
wmma_softmax_scatter_kernel(const float* __restrict__ tvals,
                            const int*   __restrict__ tids,
                            const long long* __restrict__ input_ids,
                            const float* __restrict__ W,
                            const float* __restrict__ bias,
                            const int*   __restrict__ mask_p,
                            float* __restrict__ out, int V, int R) {
  const int group = blockIdx.x;          // 16 positions per wave
  const int lane  = threadIdx.x;         // 0..31 (wave32)
  const int nn    = lane & 15;
  const int hf    = lane >> 4;
  const int mask_id = *mask_p;

  // A fragment: 16-bit A 16x32 layout -> lane holds M = lane&15,
  // element j (0..7) <-> K = hf*8 + j ; elements 8..15 <-> K = 16.. (zero pad)
  int arow = group * 16 + nn;
  if (arow >= R) arow = R - 1;           // clamp (guarded rows never scatter)
  v16h a, bf;
#pragma unroll
  for (int j = 0; j < 8; ++j) {
    a[j]      = (_Float16)tvals[arow * KTOP + hf * 8 + j];
    a[j + 8]  = (_Float16)0.f;
    // B fragment (mirror layout, N = lane&15): B[k][n] = W[n][k] = W[n*16 + k]
    bf[j]     = (_Float16)W[nn * 16 + hf * 8 + j];
    bf[j + 8] = (_Float16)0.f;
  }

  v8f c = {};
  // D = A x B + 0 : one 16x16x32 f16 WMMA covers all 16 positions (K padded)
  c = __builtin_amdgcn_wmma_f32_16x16x32_f16(false, a, false, bf,
                                             (short)0, c, false, false);

  const float bn = bias[nn];
#pragma unroll
  for (int vv = 0; vv < 8; ++vv) {
    // C/D layout: VGPR vv, lanes 0-15 -> M = vv ; lanes 16-31 -> M = vv + 8 ; N = lane&15
    const int m   = vv + hf * 8;
    const int row = group * 16 + m;
    float x = c[vv] + bn;

    float mx = x;                         // softmax over the 16 N-slots
#pragma unroll
    for (int off = 1; off < 16; off <<= 1) mx = fmaxf(mx, __shfl_xor(mx, off, 32));
    const float e = __expf(x - mx);
    float s = e;
#pragma unroll
    for (int off = 1; off < 16; off <<= 1) s += __shfl_xor(s, off, 32);
    const float o = e / s;

    if (row < R && input_ids[row] == (long long)mask_id) {
      const int idx = tids[row * KTOP + nn];
      out[(size_t)row * (size_t)V + (size_t)idx] = o;   // RT: line is L2-resident
    }
  }
}

// ---------------------------------------------------------------------------

extern "C" void kernel_launch(void* const* d_in, const int* in_sizes, int n_in,
                              void* d_out, int out_size, void* d_ws, size_t ws_size,
                              hipStream_t stream) {
  const float*     logits    = (const float*)d_in[0];
  const long long* input_ids = (const long long*)d_in[1];
  const float*     W         = (const float*)d_in[2];
  const float*     bias      = (const float*)d_in[3];
  const int*       mask_p    = (const int*)d_in[4];   // scalar on device (LE: low 32b ok)

  const int R = in_sizes[1];                // B*S rows
  const int V = in_sizes[0] / R;            // vocab width

  float* tvals = (float*)d_ws;
  int*   tids  = (int*)((char*)d_ws + (size_t)R * KTOP * sizeof(float));

  // 1) zero the dense output (NT b128 store stream)
  zero_out_kernel<<<4096, 256, 0, stream>>>((float*)d_out, (size_t)out_size);

  // 2) per-row top-16 (NT b128 load stream + register top-k)
  topk_rows_kernel<<<R, 256, 0, stream>>>(logits, tvals, tids, V);

  // 3) 16x16 linear via WMMA + softmax + masked scatter
  const int groups = (R + 15) / 16;
  wmma_softmax_scatter_kernel<<<groups, 32, 0, stream>>>(
      tvals, tids, input_ids, W, bias, mask_p, (float*)d_out, V, R);
}
